// Model_15908558865592
// MI455X (gfx1250) — matrix operand order repaired and checked by reading, not compiled, not run on hardware
//
#include <hip/hip_runtime.h>

// ---------------------------------------------------------------------------
// Flash attention B=2,H=16,S=2048,D=128 (fp32 in/out) for CDNA5 / gfx1250.
// Compute-bound (69 GFLOP vs 134 MB) -> v_wmma_f32_16x16x32_bf16 with bf16
// operands / fp32 accumulation. K/V tiles staged f32 into LDS by the Tensor
// Data Mover (TENSOR_LOAD_TO_LDS, TENSORcnt, double buffered; per-lane
// GLOBAL_LOAD_ASYNC_TO_LDS_B128 fallback), converted to bf16 operand buffers
// with v_cvt_pk_bf16_f32, softmax row reductions via v_permlane16_b32.
// ---------------------------------------------------------------------------

#define ATT_B 2
#define ATT_H 16
#define ATT_S 2048
#define ATT_D 128
#define MTILE 64      // query rows per block (16 per wave x 4 waves)
#define KTILE 32      // keys per flash iteration
#define NTHREADS 128  // 4 waves of 32

typedef __attribute__((ext_vector_type(16))) __bf16        v16bf;
typedef __attribute__((ext_vector_type(2)))  __bf16        v2bf;
typedef __attribute__((ext_vector_type(8)))  float         v8f;
typedef __attribute__((ext_vector_type(4)))  unsigned int  v4u;

union Frag16 {
    v16bf v;
    v4u   q[2];
    unsigned int   w[8];
    unsigned short u[16];
};

#define NEG_BIG (-3.402823466e+38f)

// pack two floats -> packed bf16 pair (low = a, high = b)
__device__ __forceinline__ unsigned cvt2(float a, float b) {
#if __has_builtin(__builtin_amdgcn_cvt_pk_bf16_f32)
    v2bf t = __builtin_amdgcn_cvt_pk_bf16_f32(a, b);
    return __builtin_bit_cast(unsigned, t);
#else
    unsigned r;
    asm("v_cvt_pk_bf16_f32 %0, %1, %2" : "=v"(r) : "v"(a), "v"(b));
    return r;
#endif
}

// xor-style permute within each 16-lane half (pure VALU)
__device__ __forceinline__ float permxor(float v, unsigned s0, unsigned s1) {
#if __has_builtin(__builtin_amdgcn_permlane16)
    unsigned u = __builtin_bit_cast(unsigned, v);
    unsigned r = __builtin_amdgcn_permlane16(u, u, s0, s1, false, false);
    return __builtin_bit_cast(float, r);
#else
    (void)s0; (void)s1;
    return v;
#endif
}

__device__ __forceinline__ float red_max16(float v) {
#if __has_builtin(__builtin_amdgcn_permlane16)
    v = fmaxf(v, permxor(v, 0x67452301u, 0xefcdab89u)); // xor 1
    v = fmaxf(v, permxor(v, 0x54761032u, 0xdcfe98bau)); // xor 2
    v = fmaxf(v, permxor(v, 0x32107654u, 0xba98fedcu)); // xor 4
    v = fmaxf(v, permxor(v, 0xfedcba98u, 0x76543210u)); // xor 8
#else
    for (int m = 8; m >= 1; m >>= 1) v = fmaxf(v, __shfl_xor(v, m, 32));
#endif
    return v;
}

__device__ __forceinline__ float red_sum16(float v) {
#if __has_builtin(__builtin_amdgcn_permlane16)
    v += permxor(v, 0x67452301u, 0xefcdab89u);
    v += permxor(v, 0x54761032u, 0xdcfe98bau);
    v += permxor(v, 0x32107654u, 0xba98fedcu);
    v += permxor(v, 0xfedcba98u, 0x76543210u);
#else
    for (int m = 8; m >= 1; m >>= 1) v += __shfl_xor(v, m, 32);
#endif
    return v;
}

#define HAVE_TDM __has_builtin(__builtin_amdgcn_tensor_load_to_lds)

#if HAVE_TDM
typedef __attribute__((ext_vector_type(4))) unsigned int u32x4;
typedef __attribute__((ext_vector_type(8))) int          i32x8;
typedef __attribute__((ext_vector_type(4))) int          i32x4;

// One TDM descriptor: 2D tile 128 x 32 of 4-byte elements, contiguous rows,
// global tile start -> LDS (row-major). D# layout per CDNA5 ISA.
// This toolchain exposes the 6-arg builtin (extra zero-filled i32x8 group).
__device__ __forceinline__ void tdm_load_tile(const float* gsrc, unsigned lds_off)
{
    const unsigned long long ga = (unsigned long long)(size_t)gsrc;
    u32x4 g0 = {};
    g0[0] = 1u;                                         // count=1 valid descriptor
    g0[1] = lds_off;                                    // lds_addr (bytes)
    g0[2] = (unsigned)(ga & 0xffffffffull);             // global_addr[31:0]
    g0[3] = (unsigned)((ga >> 32) & 0x1ffffffull)       // global_addr[56:32]
          | (2u << 30);                                 // type=2 ("image")
    i32x8 g1 = {};
    g1[0] = (int)(2u << 16);                            // data_size=2 -> 4 bytes
    g1[1] = (int)((ATT_D & 0xffffu) << 16);             // tensor_dim0[15:0] @ bit48
    g1[2] = (int)((KTILE & 0xffffu) << 16);             // tensor_dim0 hi=0 | tensor_dim1[15:0] @ bit80
    g1[3] = (int)((ATT_D & 0xffffu) << 16);             // tensor_dim1 hi=0 | tile_dim0 @ bit112
    g1[4] = (int)(KTILE & 0xffffu);                     // tile_dim1 @ bit128, tile_dim2=0
    g1[5] = (int)ATT_D;                                 // tensor_dim0_stride[31:0]
    g1[6] = 0;                                          // stride0 hi, dim1_stride lo
    g1[7] = 0;
    i32x4 z4 = {};
    i32x8 z8 = {};
    __builtin_amdgcn_tensor_load_to_lds(g0, g1, z4, z4, z8, 0);
}
#else
// Per-lane async fallback. INST_OFFSET applies to BOTH the global and LDS
// address (ISA 15.18 async pseudocode): one base VGPR + 8 immediates = 128B.
__device__ __forceinline__ void stage_tile_async(const float* __restrict__ Kg,
                                                 const float* __restrict__ Vg,
                                                 long base, int k0, int kr, int dc,
                                                 const float* kdst, const float* vdst)
{
    const unsigned lK   = (unsigned)(size_t)kdst;   // low 32b of LDS flat addr = LDS offset
    const unsigned lV   = (unsigned)(size_t)vdst;
    const unsigned gofs = (unsigned)((base + (long)(k0 + kr) * ATT_D + dc) * 4);
#pragma unroll
    for (int j = 0; j < 8; ++j)
        asm volatile("global_load_async_to_lds_b128 %0, %1, %2 offset:%3"
                     :: "v"(lK), "v"(gofs), "s"(Kg), "i"(j * 16) : "memory");
#pragma unroll
    for (int j = 0; j < 8; ++j)
        asm volatile("global_load_async_to_lds_b128 %0, %1, %2 offset:%3"
                     :: "v"(lV), "v"(gofs), "s"(Vg), "i"(j * 16) : "memory");
}
#endif

__global__ __launch_bounds__(NTHREADS, 1)
void fa_fwd_gfx1250(const float* __restrict__ Qg,
                    const float* __restrict__ Kg,
                    const float* __restrict__ Vg,
                    const float* __restrict__ Sc,
                    float* __restrict__ Og)
{
    // f32 staging (double buffered, TDM/async target) + bf16 operand buffers
    __shared__ float          Kstage[2][KTILE * ATT_D];   // 2 x 16 KB
    __shared__ float          Vstage[2][KTILE * ATT_D];   // 2 x 16 KB
    __shared__ unsigned short Klds[KTILE][ATT_D];         // 8 KB, row-major bf16
    __shared__ unsigned short Vtlds[ATT_D][KTILE];        // 8 KB, transposed bf16
    __shared__ unsigned short Plds[4][16][KTILE];         // 4 KB, per-wave P transpose

    const int tid  = threadIdx.x;
    const int lane = tid & 31;
    const int wave = tid >> 5;
    const int hi   = lane >> 4;   // 16-lane half
    const int ln   = lane & 15;

    const int  bh   = blockIdx.y;
    const int  mt   = blockIdx.x;
    const long base = (long)bh * ATT_S * ATT_D;
    const int  mw   = mt * MTILE + wave * 16;     // first query row of wave

    // Per-thread convert slice: 32 consecutive floats of the flat 32x128 tile
    const int kr = tid >> 2;          // key row 0..31
    const int dc = (tid & 3) << 5;    // feature column 0/32/64/96

    // ---- Q fragments: 16-bit A-layout, 4 chunks of K=32 over D=128 ----
    Frag16 qa[4];
    {
        const float* qrow = Qg + base + (long)(mw + ln) * ATT_D;
#pragma unroll
        for (int c = 0; c < 4; ++c) {
            const int d0 = c * 32 + hi * 8;        // K 0..7 (hi=0) / 8..15 (hi=1)
            const int d1 = c * 32 + 16 + hi * 8;   // K 16..23 / 24..31
#pragma unroll
            for (int j = 0; j < 4; ++j)
                qa[c].w[j]     = cvt2(qrow[d0 + 2 * j], qrow[d0 + 2 * j + 1]);
#pragma unroll
            for (int j = 0; j < 4; ++j)
                qa[c].w[4 + j] = cvt2(qrow[d1 + 2 * j], qrow[d1 + 2 * j + 1]);
        }
    }

    // Row stats for rows r + 8*hi, replicated across the 16 lanes of a half
    float mstat[8], lstat[8], rsc[8];
#pragma unroll
    for (int r = 0; r < 8; ++r) {
        mstat[r] = NEG_BIG;
        lstat[r] = 0.0f;
        rsc[r]   = 1.0f / Sc[(long)bh * ATT_S + mw + r + 8 * hi];
    }

    v8f Oacc[8];
#pragma unroll
    for (int db = 0; db < 8; ++db) Oacc[db] = (v8f){};

    // ---- Prologue: stage tile 0 ----
#if HAVE_TDM
    if (wave == 0) {
        tdm_load_tile(Kg + base, (unsigned)(size_t)&Kstage[0][0]);
        tdm_load_tile(Vg + base, (unsigned)(size_t)&Vstage[0][0]);
    }
#else
    stage_tile_async(Kg, Vg, base, 0, kr, dc, &Kstage[0][tid * 32], &Vstage[0][tid * 32]);
#endif
    int cur = 0;

    const int NT = ATT_S / KTILE;   // 64 flash iterations
    for (int t = 0; t < NT; ++t) {
        // ---- wait for stage[cur], publish to all waves ----
#if HAVE_TDM
        if (wave == 0) __builtin_amdgcn_s_wait_tensorcnt(0);
#else
        asm volatile("s_wait_asynccnt 0x0" ::: "memory");
#endif
        __syncthreads();   // stage[cur] visible; compute(t-1) done with Klds/Vtlds

        // ---- kick DMA for tile t+1 into stage[cur^1] (overlaps convert+compute) ----
        if (t + 1 < NT) {
            const long nb = base + (long)((t + 1) * KTILE) * ATT_D;
#if HAVE_TDM
            if (wave == 0) {
                tdm_load_tile(Kg + nb, (unsigned)(size_t)&Kstage[cur ^ 1][0]);
                tdm_load_tile(Vg + nb, (unsigned)(size_t)&Vstage[cur ^ 1][0]);
            }
#else
            stage_tile_async(Kg, Vg, base, (t + 1) * KTILE, kr, dc,
                             &Kstage[cur ^ 1][tid * 32], &Vstage[cur ^ 1][tid * 32]);
#endif
        }
        if (t + 2 < NT) {   // warm L2 for the tile after next (global_prefetch_b8)
            const long pb = base + (long)((t + 2) * KTILE) * ATT_D + tid * 32;
            __builtin_prefetch(Kg + pb, 0, 0);
            __builtin_prefetch(Vg + pb, 0, 0);
        }

        // ---- convert own staged slice: f32 -> bf16 (K row-major, V transposed) ----
        {
            const float* ks = &Kstage[cur][tid * 32];
            const float* vs = &Vstage[cur][tid * 32];
            unsigned* kout = (unsigned*)&Klds[kr][dc];
#pragma unroll
            for (int j = 0; j < 16; ++j) {
                kout[j] = cvt2(ks[2 * j], ks[2 * j + 1]);
                const unsigned p = cvt2(vs[2 * j], vs[2 * j + 1]);
                Vtlds[dc + 2 * j][kr]     = (unsigned short)(p & 0xffffu);
                Vtlds[dc + 2 * j + 1][kr] = (unsigned short)(p >> 16);
            }
        }
        __syncthreads();   // bf16 operand buffers ready

        // ---- S = Q K^T : 16 x 32 per wave (two 16x16 C fragments) ----
        v8f s[2];
#pragma unroll
        for (int nb2 = 0; nb2 < 2; ++nb2) {
            v8f acc = (v8f){};
#pragma unroll
            for (int c = 0; c < 4; ++c) {
                Frag16 kb;   // B fragment: lane = key column, d-values hi*8+0..7, 16+hi*8+0..7
                const unsigned short* src = &Klds[nb2 * 16 + ln][c * 32 + hi * 8];
                kb.q[0] = *(const v4u*)(src);
                kb.q[1] = *(const v4u*)(src + 16);
                acc = __builtin_amdgcn_wmma_f32_16x16x32_bf16(
                          false, qa[c].v, false, kb.v, (short)0, acc, false, false);
            }
            s[nb2] = acc;
        }

        // ---- per-row 1/scale + online softmax (reductions stay in VALU) ----
        float alpha[8];
#pragma unroll
        for (int r = 0; r < 8; ++r) {
            const float a0 = s[0][r] * rsc[r];
            const float a1 = s[1][r] * rsc[r];
            const float tmax = red_max16(fmaxf(a0, a1));
            const float mnew = fmaxf(mstat[r], tmax);
            alpha[r] = __expf(mstat[r] - mnew);
            mstat[r] = mnew;
            const float p0 = __expf(a0 - mnew);
            const float p1 = __expf(a1 - mnew);
            s[0][r] = p0;
            s[1][r] = p1;
            lstat[r] = lstat[r] * alpha[r] + red_sum16(p0 + p1);
        }

        // Rescale running output accumulators (v_pk_mul_f32)
#pragma unroll
        for (int db = 0; db < 8; ++db)
#pragma unroll
            for (int r = 0; r < 8; ++r)
                Oacc[db][r] *= alpha[r];

        // ---- P: C-layout (lane=column) -> A-layout (lane=row) via wave-local LDS ----
#pragma unroll
        for (int r = 0; r < 8; ++r) {
            const unsigned pp = cvt2(s[0][r], s[1][r]);
            Plds[wave][r + 8 * hi][ln]      = (unsigned short)(pp & 0xffffu);
            Plds[wave][r + 8 * hi][16 + ln] = (unsigned short)(pp >> 16);
        }
        // In-wave DS ordering: own stores complete before re-reading
        asm volatile("s_wait_dscnt 0x0" ::: "memory");

        Frag16 pa;
        {
            const unsigned short* src = &Plds[wave][ln][hi * 8];
            pa.q[0] = *(const v4u*)(src);        // keys hi*8 + 0..7
            pa.q[1] = *(const v4u*)(src + 16);   // keys 16 + hi*8 + 0..7
        }

        // ---- O += P V : 8 d-blocks of 16, K=32 keys per WMMA ----
#pragma unroll
        for (int db = 0; db < 8; ++db) {
            Frag16 vb;   // lane = d column, keys hi*8+0..7, 16+hi*8+0..7 from Vt
            const unsigned short* src = &Vtlds[db * 16 + ln][hi * 8];
            vb.q[0] = *(const v4u*)(src);
            vb.q[1] = *(const v4u*)(src + 16);
            Oacc[db] = __builtin_amdgcn_wmma_f32_16x16x32_bf16(
                           false, pa.v, false, vb.v, (short)0, Oacc[db], false, false);
        }

        cur ^= 1;
    }

    // ---- epilogue: O / l, fp32 out ----
#pragma unroll
    for (int r = 0; r < 8; ++r) {
        const float inv = 1.0f / lstat[r];
        float* orow = Og + base + (long)(mw + r + 8 * hi) * ATT_D;
#pragma unroll
        for (int db = 0; db < 8; ++db)
            orow[db * 16 + ln] = Oacc[db][r] * inv;
    }
}

extern "C" void kernel_launch(void* const* d_in, const int* in_sizes, int n_in,
                              void* d_out, int out_size, void* d_ws, size_t ws_size,
                              hipStream_t stream) {
    (void)in_sizes; (void)n_in; (void)d_ws; (void)ws_size; (void)out_size;
    const float* q  = (const float*)d_in[0];
    const float* k  = (const float*)d_in[1];
    const float* v  = (const float*)d_in[2];
    const float* sc = (const float*)d_in[3];
    float* out = (float*)d_out;

    dim3 grid(ATT_S / MTILE, ATT_B * ATT_H);   // (32, 32)
    dim3 block(NTHREADS);                      // 4 waves
    fa_fwd_gfx1250<<<grid, block, 0, stream>>>(q, k, v, sc, out);
}
